// E_GCL_86895778333135
// MI455X (gfx1250) — matrix operand order repaired and checked by reading, not compiled, not run on hardware
//
#include <hip/hip_runtime.h>
#include <hip/hip_bf16.h>
#include <hip/hip_fp16.h>

typedef __attribute__((ext_vector_type(16))) _Float16 v16h;
typedef __attribute__((ext_vector_type(8)))  _Float16 v8h;
typedef __attribute__((ext_vector_type(8)))  float    v8f;

#define F_DIM 64
#define H_DIM 64
#define LDS_STRIDE 80   // halves per row; 160B rows keep 16B alignment for b128 LDS loads

__device__ __forceinline__ float silu_f(float x) {
    return x / (1.0f + __expf(-x));
}

// CDNA5 split-counter wait: all outstanding DS ops complete. Used for
// intra-wave LDS staging (wave32 lockstep; no workgroup barrier needed).
__device__ __forceinline__ void lds_fence() {
    asm volatile("s_wait_dscnt 0" ::: "memory");
}

__device__ __forceinline__ v8f wmma16(v16h a, v16h b, v8f c) {
    // D = A(16x32 f16) x B(32x16 f16) + C(16x16 f32)
    return __builtin_amdgcn_wmma_f32_16x16x32_f16(
        /*neg_a=*/false, a, /*neg_b=*/false, b,
        /*c_mod=*/(short)0, c, /*reuse_a=*/false, /*reuse_b=*/false);
}

// A fragment (ISA 16-bit A 16x32 layout) from a row-major f16 row: two 16B loads.
// half==0 lanes hold K = kbase+{0..7, 16..23}; half==1 lanes hold +8.
__device__ __forceinline__ v16h a_frag_f16row(const _Float16* __restrict__ row,
                                              int kbase, int half) {
    const _Float16* p0 = row + kbase + half * 8;
    v8h lo = *(const v8h*)(p0);
    v8h hi = *(const v8h*)(p0 + 16);
    v16h a;
#pragma unroll
    for (int i = 0; i < 8; ++i) { a[i] = lo[i]; a[8 + i] = hi[i]; }
    return a;
}

// Same A fragment but from a row-major f32 row (used for the f32 `agg` tensor).
__device__ __forceinline__ v16h a_frag_f32row(const float* __restrict__ row,
                                              int kbase, int half) {
    const float* p0 = row + kbase + half * 8;
    float4 x0 = *(const float4*)(p0 + 0);
    float4 x1 = *(const float4*)(p0 + 4);
    float4 y0 = *(const float4*)(p0 + 16);
    float4 y1 = *(const float4*)(p0 + 20);
    v16h a;
    a[0]  = (_Float16)x0.x; a[1]  = (_Float16)x0.y; a[2]  = (_Float16)x0.z; a[3]  = (_Float16)x0.w;
    a[4]  = (_Float16)x1.x; a[5]  = (_Float16)x1.y; a[6]  = (_Float16)x1.z; a[7]  = (_Float16)x1.w;
    a[8]  = (_Float16)y0.x; a[9]  = (_Float16)y0.y; a[10] = (_Float16)y0.z; a[11] = (_Float16)y0.w;
    a[12] = (_Float16)y1.x; a[13] = (_Float16)y1.y; a[14] = (_Float16)y1.z; a[15] = (_Float16)y1.w;
    return a;
}

// A fragment from an f16 [16][LDS_STRIDE] staging tile in LDS.
__device__ __forceinline__ v16h a_frag_lds(const _Float16* __restrict__ m,
                                           int kc, int lane) {
    const int row = lane & 15, half = lane >> 4;
    const _Float16* p0 = m + row * LDS_STRIDE + kc * 32 + half * 8;
    v8h lo = *(const v8h*)(p0);
    v8h hi = *(const v8h*)(p0 + 16);
    v16h a;
#pragma unroll
    for (int i = 0; i < 8; ++i) { a[i] = lo[i]; a[8 + i] = hi[i]; }
    return a;
}

// B fragment (ISA 16-bit B KxN layout): lane holds column N = ntile*16 + lane%16,
// K = kbase + (lane/16)*8 + {0..7, 16..23} (same K order as the A fragment).
// Weights pre-transposed to W^T[n][kpad] f16 so this is two 16B loads.
__device__ __forceinline__ v16h b_frag(const _Float16* __restrict__ wt, int kpad,
                                       int ntile, int kbase, int lane) {
    const _Float16* p = wt + (size_t)(ntile * 16 + (lane & 15)) * kpad
                           + kbase + ((lane >> 4) * 8);
    v8h lo = *(const v8h*)(p);
    v8h hi = *(const v8h*)(p + 16);
    v16h b;
#pragma unroll
    for (int i = 0; i < 8; ++i) { b[i] = lo[i]; b[8 + i] = hi[i]; }
    return b;
}

// ---------------- weight prep: W[k][n] f32 -> W^T[n][kpad] f16 (zero padded)
__global__ void prep_wt(const float* __restrict__ W, _Float16* __restrict__ out,
                        int K, int Kpad, int Nout) {
    int i = blockIdx.x * blockDim.x + threadIdx.x;
    if (i >= Kpad * Nout) return;
    int n = i / Kpad, k = i % Kpad;
    out[i] = (k < K) ? (_Float16)W[(size_t)k * Nout + n] : (_Float16)0.0f;
}

// ---------------- feature prep: h f32 -> f16 once (amortizes the ~64x gather reuse)
__global__ void prep_f16(const float* __restrict__ src, _Float16* __restrict__ dst,
                         long long n) {
    long long i = (long long)blockIdx.x * blockDim.x + threadIdx.x;
    if (i < n) dst[i] = (_Float16)src[i];
}

__global__ void zero_f32(float* __restrict__ p, long long n) {
    long long i = (long long)blockIdx.x * blockDim.x + threadIdx.x;
    if (i < n) p[i] = 0.0f;
}

// ---------------- edge kernel: one wave32 per 16-edge tile ----------------
__global__ __launch_bounds__(256) void egcl_edge(
    const _Float16* __restrict__ h16, const float* __restrict__ coord,
    const long long* __restrict__ erow, const long long* __restrict__ ecol,
    const float* __restrict__ emask,
    const _Float16* __restrict__ wWe1, const float* __restrict__ be1,
    const _Float16* __restrict__ wWe2, const float* __restrict__ be2,
    const _Float16* __restrict__ wWc1, const float* __restrict__ bc1,
    const float* __restrict__ Wc2,
    float* __restrict__ agg, float* __restrict__ aggc, float* __restrict__ cnt,
    int nE)
{
    __shared__ _Float16 sM[8][16 * LDS_STRIDE];
    __shared__ float    sMask[8][16];
    __shared__ float    sC[8][16];
    __shared__ int      sRow[8][16];

    const int wave = threadIdx.x >> 5;
    const int lane = threadIdx.x & 31;
    const int half = lane >> 4;   // 0: M rows 0..7 in C, 1: M rows 8..15
    const int lm   = lane & 15;   // N within tile for B/C; edge id for gather

    const long long tile = ((long long)blockIdx.x * 8 + wave) * 16;
    if (tile >= nE) return;

    long long e = tile + lm;
    const bool valid = (e < nE);
    if (!valid) e = nE - 1;

    const int r = (int)erow[e];
    const int c = (int)ecol[e];

    float dx = 0.f, dy = 0.f, dz = 0.f, radial = 0.f, msk = 0.f;
    if (half == 0) {
        dx = coord[(size_t)r * 3 + 0] - coord[(size_t)c * 3 + 0];
        dy = coord[(size_t)r * 3 + 1] - coord[(size_t)c * 3 + 1];
        dz = coord[(size_t)r * 3 + 2] - coord[(size_t)c * 3 + 2];
        radial = dx * dx + dy * dy + dz * dz;
        msk = valid ? emask[e] : 0.0f;
        sMask[wave][lm] = msk;
        sRow[wave][lm]  = r;
    }
    lds_fence();

    const _Float16* hr = h16 + (size_t)r * F_DIM;
    const _Float16* hc = h16 + (size_t)c * F_DIM;

    // ---- edge MLP layer 1: [h_row | h_col | radial] (K padded to 160) -> 64
    v8f acc[4];
#pragma unroll
    for (int t = 0; t < 4; ++t) {
        float b = be1[t * 16 + lm];
#pragma unroll
        for (int i = 0; i < 8; ++i) acc[t][i] = b;
    }
#pragma unroll
    for (int kc = 0; kc < 2; ++kc) {
        v16h a = a_frag_f16row(hr, kc * 32, half);
#pragma unroll
        for (int t = 0; t < 4; ++t)
            acc[t] = wmma16(a, b_frag(wWe1, 160, t, kc * 32, lane), acc[t]);
    }
#pragma unroll
    for (int kc = 0; kc < 2; ++kc) {
        v16h a = a_frag_f16row(hc, kc * 32, half);
#pragma unroll
        for (int t = 0; t < 4; ++t)
            acc[t] = wmma16(a, b_frag(wWe1, 160, t, 64 + kc * 32, lane), acc[t]);
    }
    {   // radial occupies padded input row 128 (K-chunk base 128, K=0 of chunk)
        v16h a;
#pragma unroll
        for (int i = 0; i < 16; ++i) a[i] = (_Float16)0.0f;
        if (half == 0) a[0] = (_Float16)radial;
#pragma unroll
        for (int t = 0; t < 4; ++t)
            acc[t] = wmma16(a, b_frag(wWe1, 160, t, 128, lane), acc[t]);
    }
    // SiLU, stage as f16 A-tile in LDS (C layout -> A layout transpose)
#pragma unroll
    for (int t = 0; t < 4; ++t)
#pragma unroll
        for (int v = 0; v < 8; ++v) {
            int M = v + 8 * half;
            sM[wave][M * LDS_STRIDE + t * 16 + lm] = (_Float16)silu_f(acc[t][v]);
        }
    lds_fence();

    // ---- edge MLP layer 2: 64 -> 64, SiLU, * edge_mask ----
    v8f macc[4];
#pragma unroll
    for (int t = 0; t < 4; ++t) {
        float b = be2[t * 16 + lm];
#pragma unroll
        for (int i = 0; i < 8; ++i) macc[t][i] = b;
    }
#pragma unroll
    for (int kc = 0; kc < 2; ++kc) {
        v16h a = a_frag_lds(&sM[wave][0], kc, lane);
#pragma unroll
        for (int t = 0; t < 4; ++t)
            macc[t] = wmma16(a, b_frag(wWe2, 64, t, kc * 32, lane), macc[t]);
    }
#pragma unroll
    for (int t = 0; t < 4; ++t)
#pragma unroll
        for (int v = 0; v < 8; ++v)
            macc[t][v] = silu_f(macc[t][v]) * sMask[wave][v + 8 * half];
    lds_fence();   // all sM reads done before overwrite

    // stage m as next A operand
#pragma unroll
    for (int t = 0; t < 4; ++t)
#pragma unroll
        for (int v = 0; v < 8; ++v) {
            int M = v + 8 * half;
            sM[wave][M * LDS_STRIDE + t * 16 + lm] = (_Float16)macc[t][v];
        }
    lds_fence();

    // ---- coord MLP: p = silu(m @ Wc1 + bc1); c = p @ Wc2 ----
    v8f pacc[4];
#pragma unroll
    for (int t = 0; t < 4; ++t) {
        float b = bc1[t * 16 + lm];
#pragma unroll
        for (int i = 0; i < 8; ++i) pacc[t][i] = b;
    }
#pragma unroll
    for (int kc = 0; kc < 2; ++kc) {
        v16h a = a_frag_lds(&sM[wave][0], kc, lane);
#pragma unroll
        for (int t = 0; t < 4; ++t)
            pacc[t] = wmma16(a, b_frag(wWc1, 64, t, kc * 32, lane), pacc[t]);
    }
    // 64 -> 1 head: per-lane partial dot, then xor-butterfly over the 16-lane group
    float part[8];
#pragma unroll
    for (int v = 0; v < 8; ++v) part[v] = 0.0f;
#pragma unroll
    for (int t = 0; t < 4; ++t) {
        float w2 = Wc2[t * 16 + lm];
#pragma unroll
        for (int v = 0; v < 8; ++v) part[v] += silu_f(pacc[t][v]) * w2;
    }
#pragma unroll
    for (int off = 1; off < 16; off <<= 1)
#pragma unroll
        for (int v = 0; v < 8; ++v) part[v] += __shfl_xor(part[v], off, 32);
    if (lm == 0) {
#pragma unroll
        for (int v = 0; v < 8; ++v) sC[wave][v + 8 * half] = part[v];
    }
    lds_fence();

    // ---- coord aggregation (lanes 0-15 own their edge) ----
    if (half == 0 && valid) {
        float cc = sC[wave][lm] * msk;
        atomicAdd(&aggc[(size_t)r * 3 + 0], dx * cc);
        atomicAdd(&aggc[(size_t)r * 3 + 1], dy * cc);
        atomicAdd(&aggc[(size_t)r * 3 + 2], dz * cc);
        atomicAdd(&cnt[r], 1.0f);
    }

    // ---- message aggregation: agg[row] += m ----
#pragma unroll
    for (int t = 0; t < 4; ++t)
#pragma unroll
        for (int v = 0; v < 8; ++v) {
            int M = v + 8 * half;
            int rowM = sRow[wave][M];
            atomicAdd(&agg[(size_t)rowM * H_DIM + t * 16 + lm], macc[t][v]);
        }
}

// ---------------- node kernel: one wave32 per 16-node tile ----------------
__global__ __launch_bounds__(256) void egcl_node(
    const _Float16* __restrict__ h16, const float* __restrict__ agg,
    const _Float16* __restrict__ wWn1, const float* __restrict__ bn1,
    const _Float16* __restrict__ wWn2, const float* __restrict__ bn2,
    float* __restrict__ hout, int nN)
{
    __shared__ _Float16 sM[8][16 * LDS_STRIDE];

    const int wave = threadIdx.x >> 5;
    const int lane = threadIdx.x & 31;
    const int half = lane >> 4;
    const int lm   = lane & 15;

    const int tile = (blockIdx.x * 8 + wave) * 16;
    if (tile >= nN) return;

    int node = tile + lm;
    if (node >= nN) node = nN - 1;
    const _Float16* ha = h16 + (size_t)node * F_DIM;
    const float*    ga = agg + (size_t)node * H_DIM;

    // layer 1: [h | agg] (K=128) -> 64, SiLU
    v8f acc[4];
#pragma unroll
    for (int t = 0; t < 4; ++t) {
        float b = bn1[t * 16 + lm];
#pragma unroll
        for (int i = 0; i < 8; ++i) acc[t][i] = b;
    }
#pragma unroll
    for (int kc = 0; kc < 2; ++kc) {
        v16h a = a_frag_f16row(ha, kc * 32, half);
#pragma unroll
        for (int t = 0; t < 4; ++t)
            acc[t] = wmma16(a, b_frag(wWn1, 128, t, kc * 32, lane), acc[t]);
    }
#pragma unroll
    for (int kc = 0; kc < 2; ++kc) {
        v16h a = a_frag_f32row(ga, kc * 32, half);
#pragma unroll
        for (int t = 0; t < 4; ++t)
            acc[t] = wmma16(a, b_frag(wWn1, 128, t, 64 + kc * 32, lane), acc[t]);
    }
#pragma unroll
    for (int t = 0; t < 4; ++t)
#pragma unroll
        for (int v = 0; v < 8; ++v) {
            int M = v + 8 * half;
            sM[wave][M * LDS_STRIDE + t * 16 + lm] = (_Float16)silu_f(acc[t][v]);
        }
    lds_fence();

    // layer 2: 64 -> 64 (+ bias, no activation)
    v8f o[4];
#pragma unroll
    for (int t = 0; t < 4; ++t) {
        float b = bn2[t * 16 + lm];
#pragma unroll
        for (int i = 0; i < 8; ++i) o[t][i] = b;
    }
#pragma unroll
    for (int kc = 0; kc < 2; ++kc) {
        v16h a = a_frag_lds(&sM[wave][0], kc, lane);
#pragma unroll
        for (int t = 0; t < 4; ++t)
            o[t] = wmma16(a, b_frag(wWn2, 64, t, kc * 32, lane), o[t]);
    }
#pragma unroll
    for (int t = 0; t < 4; ++t)
#pragma unroll
        for (int v = 0; v < 8; ++v) {
            int M = v + 8 * half;
            int nd = tile + M;
            if (nd < nN) hout[(size_t)nd * F_DIM + t * 16 + lm] = o[t][v];
        }
}

__global__ void egcl_coord_fin(const float* __restrict__ coord,
                               const float* __restrict__ aggc,
                               const float* __restrict__ cnt,
                               float* __restrict__ out, int nN) {
    int i = blockIdx.x * blockDim.x + threadIdx.x;
    if (i >= nN) return;
    float ct = fmaxf(cnt[i], 1.0f);
    out[(size_t)i * 3 + 0] = coord[(size_t)i * 3 + 0] + aggc[(size_t)i * 3 + 0] / ct;
    out[(size_t)i * 3 + 1] = coord[(size_t)i * 3 + 1] + aggc[(size_t)i * 3 + 1] / ct;
    out[(size_t)i * 3 + 2] = coord[(size_t)i * 3 + 2] + aggc[(size_t)i * 3 + 2] / ct;
}

extern "C" void kernel_launch(void* const* d_in, const int* in_sizes, int n_in,
                              void* d_out, int out_size, void* d_ws, size_t ws_size,
                              hipStream_t stream) {
    const float*     h     = (const float*)d_in[0];
    const float*     coord = (const float*)d_in[1];
    const long long* eidx  = (const long long*)d_in[2];
    const float*     emask = (const float*)d_in[3];
    const float* We1 = (const float*)d_in[4];  const float* be1 = (const float*)d_in[5];
    const float* We2 = (const float*)d_in[6];  const float* be2 = (const float*)d_in[7];
    const float* Wn1 = (const float*)d_in[8];  const float* bn1 = (const float*)d_in[9];
    const float* Wn2 = (const float*)d_in[10]; const float* bn2 = (const float*)d_in[11];
    const float* Wc1 = (const float*)d_in[12]; const float* bc1 = (const float*)d_in[13];
    const float* Wc2 = (const float*)d_in[14];

    const int N = in_sizes[0] / F_DIM;   // 40000
    const int E = in_sizes[3];           // 1280000
    const long long* erow = eidx;
    const long long* ecol = eidx + E;

    // workspace layout (256B-aligned slots)
    char* ws = (char*)d_ws;
    _Float16* wWe1 = (_Float16*)(ws + 0);        // 64*160*2 = 20480
    _Float16* wWe2 = (_Float16*)(ws + 20480);    // 64*64*2  =  8192
    _Float16* wWc1 = (_Float16*)(ws + 28672);    //             8192
    _Float16* wWn1 = (_Float16*)(ws + 36864);    // 64*128*2 = 16384
    _Float16* wWn2 = (_Float16*)(ws + 53248);    //             8192
    float* agg  = (float*)(ws + 61440);                    // N*64 f32
    float* aggc = agg + (size_t)N * H_DIM;                 // N*3  f32
    float* cnt  = aggc + (size_t)N * 3;                    // N    f32
    _Float16* h16 = (_Float16*)(cnt + N);                  // N*64 f16

    // 1) prep weights (f32 [K][N] -> f16 W^T[n][Kpad], zero padded)
    prep_wt<<<(160 * 64 + 255) / 256, 256, 0, stream>>>(We1, wWe1, 129, 160, 64);
    prep_wt<<<(64  * 64 + 255) / 256, 256, 0, stream>>>(We2, wWe2, 64, 64, 64);
    prep_wt<<<(64  * 64 + 255) / 256, 256, 0, stream>>>(Wc1, wWc1, 64, 64, 64);
    prep_wt<<<(128 * 64 + 255) / 256, 256, 0, stream>>>(Wn1, wWn1, 128, 128, 64);
    prep_wt<<<(64  * 64 + 255) / 256, 256, 0, stream>>>(Wn2, wWn2, 64, 64, 64);

    // 1b) prep node features once: f32 -> f16 (each row is re-gathered ~64x)
    long long nh = (long long)N * F_DIM;
    prep_f16<<<(int)((nh + 255) / 256), 256, 0, stream>>>(h, h16, nh);

    // 2) zero the accumulators (agg | aggc | cnt are contiguous: N*68 floats)
    long long nz = (long long)N * (H_DIM + 3 + 1);
    zero_f32<<<(int)((nz + 255) / 256), 256, 0, stream>>>(agg, nz);

    // 3) edge phase: 16 edges per wave, 8 waves per block
    {
        long long tiles = ((long long)E + 15) / 16;
        int blocks = (int)((tiles + 7) / 8);
        egcl_edge<<<blocks, 256, 0, stream>>>(
            h16, coord, erow, ecol, emask,
            wWe1, be1, wWe2, be2, wWc1, bc1, Wc2,
            agg, aggc, cnt, E);
    }

    // 4) node phase
    {
        int tiles = (N + 15) / 16;
        int blocks = (tiles + 7) / 8;
        egcl_node<<<blocks, 256, 0, stream>>>(
            h16, agg, wWn1, bn1, wWn2, bn2, (float*)d_out, N);
    }

    // 5) coord finalize -> second output slab
    egcl_coord_fin<<<(N + 255) / 256, 256, 0, stream>>>(
        coord, aggc, cnt, (float*)d_out + (size_t)N * F_DIM, N);
}
